// SelectiveSSM_63874753626567
// MI455X (gfx1250) — matrix operand order repaired and checked by reading, not compiled, not run on hardware
//
#include <hip/hip_runtime.h>
#include <math.h>

#define D_MODEL 1024
#define D_STATE 16
#define D_CONV  4
#define D_INNER 2048
#define DT_RANK 64
#define BATCH   2
#define SEQLEN  2048

typedef __attribute__((ext_vector_type(16))) __bf16 v16bf;
typedef __attribute__((ext_vector_type(8)))  __bf16 v8bf;
typedef __attribute__((ext_vector_type(4)))  __bf16 v4bf;
typedef __attribute__((ext_vector_type(8)))  float  v8f;
typedef __attribute__((ext_vector_type(4)))  float  v4f;

__device__ __forceinline__ float silu_f(float x) {
    return x / (1.0f + __expf(-x));
}

__device__ __forceinline__ v8bf ld8_lds(const __bf16* p) {
    return *(const v8bf*)__builtin_assume_aligned(p, 16);   // ds_load_b128
}
__device__ __forceinline__ v16bf cat8(v8bf a, v8bf b) {
    return __builtin_shufflevector(a, b, 0, 1, 2, 3, 4, 5, 6, 7,
                                         8, 9, 10, 11, 12, 13, 14, 15);
}

// ---------------------------------------------------------------------------
// GEMM: C(MxN) = A(MxK) @ B(KxN), fp32 in/out, bf16x3 WMMA core (hi*hi+hi*lo+lo*hi).
// MODE 0: C = A@B
// MODE 1: C = softplus(A@B + bias[n])
// MODE 2: A elements gated on the fly: a = A * silu(Z)
// Block: 256 threads = 8 waves. Block tile 256(M) x 64(N), K-step 32.
// Each wave owns a 32(M) x 64(N) tile (2 A-fragments x 4 B-subtiles = 8 accs)
// so every B fragment feeds 6 WMMAs (ds:wmma ~ 1:1 in steady state).
// LDS tiles stored K-contiguous (A:[m][k], B transposed:[n][k]) with padded
// stride KP=40 bf16 (80B, 16B-aligned rows) so fragments load as ds_load_b128.
// Requires M % 256 == 0, K % 32 == 0, N % 4 == 0 (true for all call sites).
// ---------------------------------------------------------------------------
template<int MODE>
__global__ __launch_bounds__(256)
void gemm_bf16x3(const float* __restrict__ A, int lda,
                 const float* __restrict__ B, int ldb,
                 float* __restrict__ C, int ldc,
                 int M, int N, int K,
                 const float* __restrict__ bias,
                 const float* __restrict__ Z, int ldz)
{
    constexpr int KP = 40;  // padded row stride in bf16 elements
    __shared__ __attribute__((aligned(16))) __bf16 sa_hi[256 * KP];
    __shared__ __attribute__((aligned(16))) __bf16 sa_lo[256 * KP];
    __shared__ __attribute__((aligned(16))) __bf16 sb_hi[64 * KP];
    __shared__ __attribute__((aligned(16))) __bf16 sb_lo[64 * KP];

    const int tid  = threadIdx.x;
    const int wave = tid >> 5;
    const int lane = tid & 31;
    const int half = lane >> 4;
    const int l16  = lane & 15;

    const int m_block = blockIdx.y * 256;
    const int n_block = blockIdx.x * 64;
    const int m_wave  = m_block + wave * 32;   // 32 rows per wave

    v8f acc[2][4] = {};

    for (int k0 = 0; k0 < K; k0 += 32) {
        // ---- A tile 256(m) x 32(k): float4 global loads, b64 LDS stores ----
        #pragma unroll
        for (int i = 0; i < 8; ++i) {
            int q  = i * 256 + tid;      // 0..2047 float4 slots
            int kq = q & 7;              // float4 index within row
            int r  = q >> 3;             // m row in tile (0..255)
            v4f a = *(const v4f*)(A + (size_t)(m_block + r) * lda + k0 + 4 * kq);
            if (MODE == 2) {
                v4f zv = *(const v4f*)(Z + (size_t)(m_block + r) * ldz + k0 + 4 * kq);
                #pragma unroll
                for (int jj = 0; jj < 4; ++jj) a[jj] *= silu_f(zv[jj]);
            }
            v4bf h, l;
            #pragma unroll
            for (int jj = 0; jj < 4; ++jj) {
                __bf16 hb = (__bf16)a[jj];
                h[jj] = hb;
                l[jj] = (__bf16)(a[jj] - (float)hb);
            }
            *(v4bf*)__builtin_assume_aligned(sa_hi + r * KP + 4 * kq, 8) = h;
            *(v4bf*)__builtin_assume_aligned(sa_lo + r * KP + 4 * kq, 8) = l;
        }
        // ---- B tile 32(k) x 64(n): float4 global loads, transposed stores ----
        #pragma unroll
        for (int i = 0; i < 2; ++i) {
            int q  = i * 256 + tid;      // 0..511 float4 slots
            int nq = q & 15;             // float4 index along n
            int r  = q >> 4;             // k row in tile
            int gn = n_block + 4 * nq;
            v4f b = {0.0f, 0.0f, 0.0f, 0.0f};
            if (gn < N) b = *(const v4f*)(B + (size_t)(k0 + r) * ldb + gn);
            #pragma unroll
            for (int jj = 0; jj < 4; ++jj) {
                __bf16 hb = (__bf16)b[jj];
                sb_hi[(4 * nq + jj) * KP + r] = hb;
                sb_lo[(4 * nq + jj) * KP + r] = (__bf16)(b[jj] - (float)hb);
            }
        }
        __syncthreads();

        // ---- two A fragments per wave (rows +0 and +16) ----
        v16bf a_hi[2], a_lo[2];
        #pragma unroll
        for (int mi = 0; mi < 2; ++mi) {
            const int arow = (wave * 32 + mi * 16 + l16) * KP + half * 8;
            a_hi[mi] = cat8(ld8_lds(sa_hi + arow), ld8_lds(sa_hi + arow + 16));
            a_lo[mi] = cat8(ld8_lds(sa_lo + arow), ld8_lds(sa_lo + arow + 16));
        }

        // ---- 4 N-subtiles; each B fragment feeds 6 WMMAs ----
        #pragma unroll
        for (int j = 0; j < 4; ++j) {
            const int brow = (j * 16 + l16) * KP + half * 8;
            v16bf b_hi = cat8(ld8_lds(sb_hi + brow), ld8_lds(sb_hi + brow + 16));
            v16bf b_lo = cat8(ld8_lds(sb_lo + brow), ld8_lds(sb_lo + brow + 16));
            #pragma unroll
            for (int mi = 0; mi < 2; ++mi) {
                acc[mi][j] = __builtin_amdgcn_wmma_f32_16x16x32_bf16(false, a_hi[mi], false, b_hi, (short)0, acc[mi][j], false, false);
                acc[mi][j] = __builtin_amdgcn_wmma_f32_16x16x32_bf16(false, a_hi[mi], false, b_lo, (short)0, acc[mi][j], false, false);
                acc[mi][j] = __builtin_amdgcn_wmma_f32_16x16x32_bf16(false, a_lo[mi], false, b_hi, (short)0, acc[mi][j], false, false);
            }
        }
        __syncthreads();
    }

    // ---- epilogue; C/D layout: VGPR r -> m = r + 8*half, n = lane&15 ----
    #pragma unroll
    for (int mi = 0; mi < 2; ++mi) {
        #pragma unroll
        for (int j = 0; j < 4; ++j) {
            #pragma unroll
            for (int r = 0; r < 8; ++r) {
                int m = m_wave + mi * 16 + r + 8 * half;
                int n = n_block + j * 16 + l16;
                if (n < N) {
                    float v = acc[mi][j][r];
                    if (MODE == 1) {
                        v += bias[n];
                        v = (v > 20.0f) ? v : log1pf(__expf(v));
                    }
                    C[(size_t)m * ldc + n] = v;
                }
            }
        }
    }
}

// ---------------------------------------------------------------------------
// Depthwise causal conv1d (width 4) + bias + SiLU.
// Reads the xi half of xz (b,l,2*D_INNER); writes xi (b,l,D_INNER).
// ---------------------------------------------------------------------------
__global__ __launch_bounds__(256)
void conv_silu_kernel(const float* __restrict__ xz, const float* __restrict__ w,
                      const float* __restrict__ bias, float* __restrict__ xi)
{
    int idx = blockIdx.x * 256 + threadIdx.x;
    if (idx >= BATCH * SEQLEN * D_INNER) return;
    int d = idx % D_INNER;
    int l = (idx / D_INNER) % SEQLEN;
    int b = idx / (D_INNER * SEQLEN);
    float acc = bias[d];
    #pragma unroll
    for (int k = 0; k < D_CONV; ++k) {
        int ls = l - (D_CONV - 1) + k;
        if (ls >= 0) {
            float xv = xz[(size_t)(b * SEQLEN + ls) * (2 * D_INNER) + d];
            acc += w[d * D_CONV + k] * xv;
        }
    }
    xi[(size_t)idx] = silu_f(acc);
}

// ---------------------------------------------------------------------------
// Selective scan. One lane per (b, d, n): 65536 threads = 2048 wave32.
// y = sum_n(h*C) reduced across the 16-lane state group via __shfl_xor.
// dt buffer is overwritten in place with y (read-before-write per step).
// ---------------------------------------------------------------------------
__global__ __launch_bounds__(256)
void scan_kernel(const float* __restrict__ xi,     // (B,L,D_INNER)
                 const float* __restrict__ xdbl,   // (B,L,96)
                 float* __restrict__ dt_y,         // (B,L,D_INNER): in dt, out y
                 const float* __restrict__ A_log,  // (D_INNER,16)
                 const float* __restrict__ Dvec)   // (D_INNER)
{
    int g  = blockIdx.x * 256 + threadIdx.x;   // BATCH*D_INNER*16 threads
    int n  = g & 15;
    int ch = g >> 4;
    int d  = ch % D_INNER;
    int b  = ch / D_INNER;

    const float A  = -__expf(A_log[d * D_STATE + n]);
    const float Dd = Dvec[d];
    const int   ldx = DT_RANK + 2 * D_STATE;   // 96
    float h = 0.0f;

    for (int t = 0; t < SEQLEN; ++t) {
        size_t row = (size_t)(b * SEQLEN + t);
        float dt = dt_y[row * D_INNER + d];          // broadcast across 16 lanes
        float xv = xi[row * D_INNER + d];
        float Bt = xdbl[row * ldx + DT_RANK + n];
        float Ct = xdbl[row * ldx + DT_RANK + D_STATE + n];

        float dA = __expf(dt * A);
        h = dA * h + (dt * Bt) * xv;

        float p = h * Ct;                            // reduce over n (16 lanes)
        p += __shfl_xor(p, 1, 32);
        p += __shfl_xor(p, 2, 32);
        p += __shfl_xor(p, 4, 32);
        p += __shfl_xor(p, 8, 32);
        if (n == 0) dt_y[row * D_INNER + d] = p + Dd * xv;
    }
}

// ---------------------------------------------------------------------------
extern "C" void kernel_launch(void* const* d_in, const int* in_sizes, int n_in,
                              void* d_out, int out_size, void* d_ws, size_t ws_size,
                              hipStream_t stream)
{
    const float* x      = (const float*)d_in[0];
    const float* W_in   = (const float*)d_in[1];
    const float* conv_w = (const float*)d_in[2];
    const float* conv_b = (const float*)d_in[3];
    const float* W_x    = (const float*)d_in[4];
    const float* W_dt   = (const float*)d_in[5];
    const float* b_dt   = (const float*)d_in[6];
    const float* A_log  = (const float*)d_in[7];
    const float* Dv     = (const float*)d_in[8];
    const float* W_out  = (const float*)d_in[9];
    float* out = (float*)d_out;

    float* ws   = (float*)d_ws;
    float* xz   = ws;                                  // 4096*4096 (64 MB)
    float* xi   = xz   + (size_t)4096 * 4096;          // 4096*2048 (32 MB)
    float* xdbl = xi   + (size_t)4096 * 2048;          // 4096*96   (1.5 MB)
    float* dty  = xdbl + (size_t)4096 * 96;            // 4096*2048 (32 MB)

    const int M = BATCH * SEQLEN;                      // 4096
    dim3 blk(256);

    // 1) xz = x @ W_in                     (M=4096, K=1024, N=4096)
    gemm_bf16x3<0><<<dim3((2 * D_INNER) / 64, M / 256), blk, 0, stream>>>(
        x, D_MODEL, W_in, 2 * D_INNER, xz, 2 * D_INNER,
        M, 2 * D_INNER, D_MODEL, nullptr, nullptr, 0);

    // 2) depthwise conv + SiLU on the xi half of xz
    conv_silu_kernel<<<dim3((M * D_INNER) / 256), blk, 0, stream>>>(
        xz, conv_w, conv_b, xi);

    // 3) x_dbl = xi @ W_x                  (M=4096, K=2048, N=96)
    gemm_bf16x3<0><<<dim3(2, M / 256), blk, 0, stream>>>(
        xi, D_INNER, W_x, DT_RANK + 2 * D_STATE, xdbl, DT_RANK + 2 * D_STATE,
        M, DT_RANK + 2 * D_STATE, D_INNER, nullptr, nullptr, 0);

    // 4) dt = softplus(x_dbl[:, :64] @ W_dt + b_dt)   (M=4096, K=64, N=2048)
    gemm_bf16x3<1><<<dim3(D_INNER / 64, M / 256), blk, 0, stream>>>(
        xdbl, DT_RANK + 2 * D_STATE, W_dt, D_INNER, dty, D_INNER,
        M, D_INNER, DT_RANK, b_dt, nullptr, 0);

    // 5) selective scan; dty becomes y
    scan_kernel<<<dim3((BATCH * D_INNER * 16) / 256), blk, 0, stream>>>(
        xi, xdbl, dty, A_log, Dv);

    // 6) out = (y * silu(z)) @ W_out       (M=4096, K=2048, N=1024), z = xz[:,2048:]
    gemm_bf16x3<2><<<dim3(D_MODEL / 64, M / 256), blk, 0, stream>>>(
        dty, D_INNER, W_out, D_MODEL, out, D_MODEL,
        M, D_MODEL, D_INNER, nullptr, xz + D_INNER, 2 * D_INNER);
}
